// Recurrent_Net_3942779978287
// MI455X (gfx1250) — matrix operand order repaired and checked by reading, not compiled, not run on hardware
//
#include <hip/hip_runtime.h>
#include <cstdint>
#include <cstddef>

// Problem constants (from reference): B,T,IN,H,OUT = 512,512,128,256,128
#define B_    512
#define T_    512
#define IN_   128
#define H_    256
#define OUT_  128
#define G3_   768   // 3*H

typedef __attribute__((ext_vector_type(16))) __bf16 v16bf;
typedef __attribute__((ext_vector_type(8)))  float  v8f;

union Frag16 { uint4 q[2]; unsigned u[8]; v16bf v; };

__device__ __forceinline__ unsigned short f32_to_bf16(float f) {
  unsigned b = __builtin_bit_cast(unsigned, f);
  b += 0x7FFFu + ((b >> 16) & 1u);   // round-to-nearest-even
  return (unsigned short)(b >> 16);
}

__device__ __forceinline__ unsigned pack2_bf16(float lo, float hi) {
  unsigned lb = __builtin_bit_cast(unsigned, lo);
  unsigned hb = __builtin_bit_cast(unsigned, hi);
  lb += 0x7FFFu + ((lb >> 16) & 1u);
  hb += 0x7FFFu + ((hb >> 16) & 1u);
  return (lb >> 16) | (hb & 0xFFFF0000u);
}

// A-matrix 16x32 bf16 fragment from an LDS *bf16* tile (row stride in elems).
// Lane L: row m=L&15, k-group (L>>4)*8. Its 8 dword-pairs live at k-offsets
// {0,2,4,6} and {16,18,20,22} -> two contiguous 16-byte runs => 2x ds_load_b128.
__device__ __forceinline__ Frag16 load_a(const unsigned short* base, int stride, int lane) {
  Frag16 a;
  const unsigned short* p = base + (lane & 15) * stride + ((lane >> 4) << 3);
  a.q[0] = *(const uint4*)(p);        // k-offsets 0..7
  a.q[1] = *(const uint4*)(p + 16);   // k-offsets 16..23
  return a;
}

// B-matrix 32x16 bf16 fragment = W^T tile from row-major bf16 W[N x K].
// Lane L: column n, K elems {8h..8h+7} and {16+8h..16+8h+7} => 2x b128.
__device__ __forceinline__ Frag16 load_b(const unsigned short* W,
                                         int K, int nb, int kb, int lane) {
  Frag16 b;
  const uint4* p = (const uint4*)(W + (size_t)(nb + (lane & 15)) * K
                                    + (size_t)(kb + ((lane >> 4) << 3)));
  b.q[0] = p[0];
  b.q[1] = p[2];
  return b;
}

__global__ void cvt_bf16_kernel(const float* __restrict__ src,
                                unsigned short* __restrict__ dst, int n) {
  int i = blockIdx.x * 256 + threadIdx.x;
  if (i < n) dst[i] = f32_to_bf16(src[i]);
}

// One workgroup owns 16 batch rows for the entire recurrence (rows of the
// batch are independent, so no inter-WG sync is ever needed).
// 8 waves; wave w owns gate columns [w*96, w*96+96) = 6 WMMA N-tiles.
__global__ __launch_bounds__(256) void gru_kernel(
    const float* __restrict__ x,            // (B,T,IN) f32
    const unsigned short* __restrict__ Wih, // (768,128) bf16
    const unsigned short* __restrict__ Whh, // (768,256) bf16
    const unsigned short* __restrict__ Wlin,// (128,256) bf16
    const float* __restrict__ bias_g,       // (768,)
    const float* __restrict__ bias_n,       // (256,)
    const float* __restrict__ bias_out,     // (128,)
    float* __restrict__ out)                // (B,OUT) f32
{
  __shared__ unsigned short xbf[16 * IN_]; //  4 KB : x_t tile (bf16)
  __shared__ float          hbuf[16 * H_]; // 16 KB : hidden state master (f32)
  __shared__ unsigned short hbf [16 * H_]; //  8 KB : hidden state mirror (bf16)
  __shared__ float          gbuf[16 * G3_];// 48 KB : sig(r), sig(z), inn
  __shared__ float          hnb [16 * H_]; // 16 KB : raw hn

  const int tid   = threadIdx.x;
  const int lane  = tid & 31;              // wave32
  const int wave  = tid >> 5;              // 0..7
  const int b0    = blockIdx.x * 16;
  const int jwave = wave * 96;             // first gate column for this wave
  const int ncol  = lane & 15;             // C/D column within tile
  const int mrow  = (lane >> 4) * 8;       // C/D row base (VGPR r -> row r+mrow)

  // hoisted biases: per-tile gate bias, per-thread bias_n pairs
  float bg[6];
#pragma unroll
  for (int tt = 0; tt < 6; ++tt) bg[tt] = bias_g[jwave + tt * 16 + ncol];
  float2 bn_r[8];
#pragma unroll
  for (int it = 0; it < 8; ++it) {
    int e = (tid + it * 256) * 2;
    bn_r[it] = *(const float2*)&bias_n[e & (H_ - 1)];
  }

  for (int i = tid; i < 16 * H_; i += 256) hbuf[i] = 0.0f;
  for (int i = tid; i < (16 * H_) / 2; i += 256) ((unsigned*)hbf)[i] = 0u;
  __syncthreads();

  for (int t = 0; t < T_; ++t) {
    // ---- stage x_t (16 x 128 f32) into LDS as bf16; 2 float4 per thread ----
#pragma unroll
    for (int it = 0; it < 2; ++it) {
      int i = tid + it * 256;              // float4 chunk id, 512 total
      int m = i >> 5;                      // 32 float4 per row
      int c = i & 31;
      float4 v = ((const float4*)(x + ((size_t)(b0 + m) * T_ + t) * IN_))[c];
      uint2 pk;
      pk.x = pack2_bf16(v.x, v.y);
      pk.y = pack2_bf16(v.z, v.w);
      ((uint2*)xbf)[i] = pk;               // ds_store_b64
      if (t + 1 < T_) {                    // keep HBM one step ahead
        __builtin_prefetch(x + ((size_t)(b0 + m) * T_ + (t + 1)) * IN_ + 4 * c, 0, 1);
      }
    }
    __syncthreads();                       // x staged; prior h-update visible

    // Anti-LICM barrier: without this, the compiler hoists all loop-invariant
    // weight-fragment loads (576 dwords per lane) out of the 512-iteration time loop
    // and spills them to scratch. This keeps them as in-loop L2 reads.
    asm volatile("" ::: "memory");

    v8f ig[6] = {}, hg[6] = {};

    // ---- input gates: x_t (16x128) @ Wih^T, K-steps of 32 ----
#pragma unroll
    for (int ks = 0; ks < IN_ / 32; ++ks) {
      Frag16 a = load_a(xbf + ks * 32, IN_, lane);
#pragma unroll
      for (int tt = 0; tt < 6; ++tt) {
        Frag16 b = load_b(Wih, IN_, jwave + tt * 16, ks * 32, lane);
        ig[tt] = __builtin_amdgcn_wmma_f32_16x16x32_bf16(
            false, a.v, false, b.v, (short)0, ig[tt], false, false);
      }
    }
    // ---- hidden gates: h (16x256) @ Whh^T ----
#pragma unroll
    for (int ks = 0; ks < H_ / 32; ++ks) {
      Frag16 a = load_a(hbf + ks * 32, H_, lane);
#pragma unroll
      for (int tt = 0; tt < 6; ++tt) {
        Frag16 b = load_b(Whh, H_, jwave + tt * 16, ks * 32, lane);
        hg[tt] = __builtin_amdgcn_wmma_f32_16x16x32_bf16(
            false, a.v, false, b.v, (short)0, hg[tt], false, false);
      }
    }

    // ---- gate nonlinearity, scatter C/D fragments to LDS (wave-uniform branch) ----
#pragma unroll
    for (int tt = 0; tt < 6; ++tt) {
      const int jbase = jwave + tt * 16;
      const int j     = jbase + ncol;
      if (jbase < 2 * H_) {                // r or z columns: sigmoid(ig+hg+b)
#pragma unroll
        for (int r = 0; r < 8; ++r) {
          float v = ig[tt][r] + hg[tt][r] + bg[tt];
          gbuf[(r + mrow) * G3_ + j] = 1.0f / (1.0f + __expf(-v));
        }
      } else {                             // n columns: keep inn and hn separate
#pragma unroll
        for (int r = 0; r < 8; ++r) {
          gbuf[(r + mrow) * G3_ + j] = ig[tt][r] + bg[tt];
          hnb[(r + mrow) * H_ + (j - 2 * H_)] = hg[tt][r];
        }
      }
    }
    __syncthreads();

    // ---- h update (2 elems/iter): n = tanh(inn + r*(hn+bn)); h = n + z*(h-n) ----
#pragma unroll
    for (int it = 0; it < 8; ++it) {
      int i = tid + it * 256;              // pair id, 2048 total
      int e = i * 2;
      int m = e >> 8;
      int j = e & (H_ - 1);
      float2 rr  = *(const float2*)&gbuf[m * G3_ + j];
      float2 zz  = *(const float2*)&gbuf[m * G3_ + H_ + j];
      float2 inn = *(const float2*)&gbuf[m * G3_ + 2 * H_ + j];
      float2 hn  = *(const float2*)&hnb[e];
      float2 h   = *(const float2*)&hbuf[e];
      float n0 = tanhf(inn.x + rr.x * (hn.x + bn_r[it].x));
      float n1 = tanhf(inn.y + rr.y * (hn.y + bn_r[it].y));
      float h0 = n0 + zz.x * (h.x - n0);
      float h1 = n1 + zz.y * (h.y - n1);
      *(float2*)&hbuf[e] = make_float2(h0, h1);
      ((unsigned*)hbf)[i] = pack2_bf16(h0, h1);
    }
    __syncthreads();
  }

  // ---- final linear: h (16x256) @ Wlin^T (256x128); wave w -> output tile w ----
  v8f acc = {};
#pragma unroll
  for (int ks = 0; ks < H_ / 32; ++ks) {
    Frag16 a = load_a(hbf + ks * 32, H_, lane);
    Frag16 b = load_b(Wlin, H_, wave * 16, ks * 32, lane);
    acc = __builtin_amdgcn_wmma_f32_16x16x32_bf16(
        false, a.v, false, b.v, (short)0, acc, false, false);
  }
  const int n  = wave * 16 + ncol;
  const float bo = bias_out[n];
#pragma unroll
  for (int r = 0; r < 8; ++r) {
    out[(size_t)(b0 + r + mrow) * OUT_ + n] = acc[r] + bo;
  }
}

extern "C" void kernel_launch(void* const* d_in, const int* in_sizes, int n_in,
                              void* d_out, int out_size, void* d_ws, size_t ws_size,
                              hipStream_t stream) {
  const float* x      = (const float*)d_in[0];
  const float* wih    = (const float*)d_in[1];   // (768,128)
  const float* whh    = (const float*)d_in[2];   // (768,256)
  const float* bias_g = (const float*)d_in[3];
  const float* bias_n = (const float*)d_in[4];
  const float* wlin   = (const float*)d_in[5];   // (128,256)
  const float* bias_o = (const float*)d_in[6];
  float* out = (float*)d_out;

  char* ws = (char*)d_ws;
  unsigned short* wih_bf  = (unsigned short*)ws;                                  // 192 KB
  unsigned short* whh_bf  = (unsigned short*)(ws + (size_t)G3_ * IN_ * 2);        // 384 KB
  unsigned short* wlin_bf = (unsigned short*)(ws + (size_t)G3_ * IN_ * 2
                                                 + (size_t)G3_ * H_ * 2);         //  64 KB

  cvt_bf16_kernel<<<(G3_ * IN_ + 255) / 256, 256, 0, stream>>>(wih,  wih_bf,  G3_ * IN_);
  cvt_bf16_kernel<<<(G3_ * H_  + 255) / 256, 256, 0, stream>>>(whh,  whh_bf,  G3_ * H_);
  cvt_bf16_kernel<<<(OUT_ * H_ + 255) / 256, 256, 0, stream>>>(wlin, wlin_bf, OUT_ * H_);

  gru_kernel<<<B_ / 16, 256, 0, stream>>>(x, wih_bf, whh_bf, wlin_bf,
                                          bias_g, bias_n, bias_o, out);
}
